// LlamaAttention_74199855006132
// MI455X (gfx1250) — compile-verified, matrix operand order checked by reading, not run on hardware
//
#include <hip/hip_runtime.h>

// ---------------- problem constants ----------------
#define BATCH 2
#define SEQ   2048
#define HIDD  4096
#define NH    32
#define NKV   8
#define HD    128
#define GROUPS (NH / NKV)
#define MROWS (BATCH * SEQ)      // 4096
#define NQ    (NH * HD)          // 4096
#define NKVD  (NKV * HD)         // 1024

typedef __attribute__((ext_vector_type(16))) __bf16 v16bf;
typedef __attribute__((ext_vector_type(8)))  __bf16 v8bf;
typedef __attribute__((ext_vector_type(8)))  float  v8f;

// Load one 16x32 bf16 WMMA A/B fragment from a row-major [rows][ld] matrix.
// Per CDNA5 ISA layout: lanes 0-15 hold row (lane&15), K = {0..7, 16..23};
// lanes 16-31 hold the same rows, K = {8..15, 24..31}.
__device__ __forceinline__ v16bf load_frag(const __bf16* base, int ld) {
    int lane = threadIdx.x & 31;
    const __bf16* p = base + (size_t)(lane & 15) * ld + ((lane & 16) ? 8 : 0);
    v8bf lo = *reinterpret_cast<const v8bf*>(p);
    v8bf hi = *reinterpret_cast<const v8bf*>(p + 16);
    return __builtin_shufflevector(lo, hi, 0,1,2,3,4,5,6,7,8,9,10,11,12,13,14,15);
}

#define WMMA_BF16(A, B, C) \
    __builtin_amdgcn_wmma_f32_16x16x32_bf16(false, (A), false, (B), (short)0, (C), false, false)

// ---------------- elementwise converts ----------------
__global__ __launch_bounds__(256) void cast_f32_bf16(const float* __restrict__ src,
                                                     __bf16* __restrict__ dst, size_t n) {
    size_t i = (size_t)blockIdx.x * blockDim.x + threadIdx.x;
    if (i < n) dst[i] = (__bf16)src[i];
}

// RoPE + reshape: src [B*S, nheads*HD] f32 -> dst [B, nheads, S, HD] bf16
__global__ __launch_bounds__(256) void rope_kernel(const float* __restrict__ src,
                                                   const int* __restrict__ pos_ids,
                                                   __bf16* __restrict__ dst, int nheads) {
    size_t tid = (size_t)blockIdx.x * blockDim.x + threadIdx.x;
    size_t total = (size_t)BATCH * SEQ * nheads * (HD / 2);
    if (tid >= total) return;
    int d  = (int)(tid & 63);
    size_t r = tid >> 6;
    int hh = (int)(r % nheads);
    size_t bs = r / nheads;             // b*S + s
    int s  = (int)(bs % SEQ);
    int bb = (int)(bs / SEQ);
    float p    = (float)pos_ids[bs];
    float freq = __powf(10000.0f, -(float)(2 * d) / (float)HD);
    float ang  = p * freq;
    float c = __cosf(ang), sn = __sinf(ang);
    const float* x = src + bs * ((size_t)nheads * HD) + (size_t)hh * HD;
    float x1 = x[d], x2 = x[d + 64];
    __bf16* o = dst + (((size_t)bb * nheads + hh) * SEQ + s) * HD;
    o[d]      = (__bf16)(x1 * c - x2 * sn);
    o[d + 64] = (__bf16)(x2 * c + x1 * sn);
}

// V: [B*S, NKV*HD] f32 -> [B, NKV, S, HD] bf16
__global__ __launch_bounds__(256) void v_reshape(const float* __restrict__ src,
                                                 __bf16* __restrict__ dst) {
    size_t tid = (size_t)blockIdx.x * blockDim.x + threadIdx.x;
    size_t total = (size_t)BATCH * SEQ * NKV * HD;
    if (tid >= total) return;
    int d  = (int)(tid & (HD - 1));
    size_t r = tid >> 7;
    int hh = (int)(r % NKV);
    size_t bs = r / NKV;
    int s  = (int)(bs % SEQ);
    int bb = (int)(bs / SEQ);
    dst[(((size_t)bb * NKV + hh) * SEQ + s) * HD + d] =
        (__bf16)src[bs * (size_t)NKVD + (size_t)hh * HD + d];
}

// ---------------- GEMM: C[M,N] f32 = X[M,K]bf16 @ W[N,K]bf16^T ----------------
// Compile-time N,K so strides fold into immediate offsets; BK=64 stage with
// register double-buffering so global loads overlap the WMMA stream.
#define BMT 128
#define BNT 128
#define BKT 64

template <int N, int K>
__global__ __launch_bounds__(256) void gemm_bf16nt(const __bf16* __restrict__ X,
                                                   const __bf16* __restrict__ W,
                                                   float* __restrict__ C) {
    __shared__ __bf16 xs[BMT][BKT];
    __shared__ __bf16 wsT[BNT][BKT];
    const int tid  = threadIdx.x;
    const int lane = tid & 31, wid = tid >> 5;
    const int wm = (wid >> 1) * 32;        // wave M offset: 0..96
    const int wn = (wid & 1) * 64;         // wave N offset: 0 or 64
    const int bm = blockIdx.y * BMT;
    const int bn = blockIdx.x * BNT;

    // staging assignment: each thread moves 32 bf16 (4x uint4) per tile
    const int srow = tid >> 1;
    const int scol = (tid & 1) * 32;
    const __bf16* gx = X + (size_t)(bm + srow) * K + scol;
    const __bf16* gw = W + (size_t)(bn + srow) * K + scol;

    v8f acc[2][4] = {};
    uint4 rx[4], rw[4];

    // prologue: stage 0
    {
        const uint4* sx = reinterpret_cast<const uint4*>(gx);
        const uint4* sw = reinterpret_cast<const uint4*>(gw);
        for (int j = 0; j < 4; ++j) { rx[j] = sx[j]; rw[j] = sw[j]; }
        uint4* dx = reinterpret_cast<uint4*>(&xs[srow][scol]);
        uint4* dw = reinterpret_cast<uint4*>(&wsT[srow][scol]);
        for (int j = 0; j < 4; ++j) { dx[j] = rx[j]; dw[j] = rw[j]; }
    }
    __syncthreads();

    for (int kb = 0; kb < K; kb += BKT) {
        const bool hasNext = (kb + BKT < K);
        if (hasNext) {   // issue next stage's global loads before the WMMAs
            const uint4* sx = reinterpret_cast<const uint4*>(gx + kb + BKT);
            const uint4* sw = reinterpret_cast<const uint4*>(gw + kb + BKT);
            for (int j = 0; j < 4; ++j) { rx[j] = sx[j]; rw[j] = sw[j]; }
            if (kb + 2 * BKT < K) {   // warm GL2 two stages ahead
                __builtin_prefetch(gx + kb + 2 * BKT, 0, 1);
                __builtin_prefetch(gw + kb + 2 * BKT, 0, 1);
            }
        }
        for (int kk = 0; kk < 2; ++kk) {
            v16bf af[2], bf[4];
            for (int mt = 0; mt < 2; ++mt) af[mt] = load_frag(&xs[wm + mt * 16][kk * 32], BKT);
            for (int nt = 0; nt < 4; ++nt) bf[nt] = load_frag(&wsT[wn + nt * 16][kk * 32], BKT);
            for (int mt = 0; mt < 2; ++mt)
                for (int nt = 0; nt < 4; ++nt)
                    acc[mt][nt] = WMMA_BF16(af[mt], bf[nt], acc[mt][nt]);
        }
        if (hasNext) {
            __syncthreads();   // all waves done reading LDS
            uint4* dx = reinterpret_cast<uint4*>(&xs[srow][scol]);
            uint4* dw = reinterpret_cast<uint4*>(&wsT[srow][scol]);
            for (int j = 0; j < 4; ++j) { dx[j] = rx[j]; dw[j] = rw[j]; }
            __syncthreads();
        }
    }

    const int half = (lane >> 4) & 1;
    const int nlo  = lane & 15;
    for (int mt = 0; mt < 2; ++mt) {
        float* cb = C + (size_t)(bm + wm + mt * 16 + half * 8) * N + bn + wn + nlo;
        for (int nt = 0; nt < 4; ++nt)
            for (int i = 0; i < 8; ++i)
                cb[i * N + nt * 16] = acc[mt][nt][i];   // i*N folds to imm offset
    }
}

// ---------------- causal GQA flash attention ----------------
// Q [B,NH,S,HD] bf16, K/V [B,NKV,S,HD] bf16 -> X2 [B*S, NH*HD] bf16
// K and V^T are staged in LDS cooperatively (shared by all 4 waves), so the
// WMMA streams feed from ds_load_b128 instead of per-wave global round-trips.
__global__ __launch_bounds__(128, 1) void attn_kernel(const __bf16* __restrict__ Q,
                                                      const __bf16* __restrict__ Kc,
                                                      const __bf16* __restrict__ Vc,
                                                      __bf16* __restrict__ X2) {
    __shared__ __bf16 kt[32][HD];        // K block, row-major [key][dim]
    __shared__ __bf16 vt[HD][32];        // V^T for current 32-key block
    __shared__ __bf16 pls[4][16][32];    // per-wave P tile

    const int lane = threadIdx.x & 31;
    const int wid  = threadIdx.x >> 5;
    const int h = blockIdx.y, b = blockIdx.z;
    const int hkv = h / GROUPS;
    const int qbase = blockIdx.x * 64 + wid * 16;
    const int half = (lane >> 4) & 1;
    const int nlo  = lane & 15;

    const __bf16* qptr = Q + (((size_t)b * NH + h) * SEQ + qbase) * HD;
    v16bf qf[4];
    for (int kk = 0; kk < 4; ++kk) qf[kk] = load_frag(qptr + kk * 32, HD);

    v8f o[8] = {};
    float mrow[8], lrow[8];
    for (int i = 0; i < 8; ++i) { mrow[i] = -1e30f; lrow[i] = 0.0f; }

    const int nkb = blockIdx.x * 2 + 2;  // causal extent in 32-key blocks
    const __bf16* kbaseptr = Kc + (((size_t)b * NKV + hkv) * SEQ) * HD;
    const __bf16* vbaseptr = Vc + (((size_t)b * NKV + hkv) * SEQ) * HD;

    for (int kb = 0; kb < nkb; ++kb) {
        __syncthreads();
        {   // cooperative staging: thread -> (key, 32-dim segment)
            const int key  = threadIdx.x >> 2;
            const int dseg = (threadIdx.x & 3) * 32;
            // K: vectorized row copy into LDS
            const uint4* ksrc = reinterpret_cast<const uint4*>(
                kbaseptr + (size_t)(kb * 32 + key) * HD + dseg);
            uint4* kdst = reinterpret_cast<uint4*>(&kt[key][dseg]);
            for (int j = 0; j < 4; ++j) kdst[j] = ksrc[j];
            // V: transpose into LDS
            const __bf16* vsrc = vbaseptr + (size_t)(kb * 32 + key) * HD + dseg;
            for (int j = 0; j < 32; ++j) vt[dseg + j][key] = vsrc[j];
            if (kb + 1 < nkb) {   // warm L2 for next key block
                __builtin_prefetch(vbaseptr + (size_t)((kb + 1) * 32 + key) * HD + dseg, 0, 1);
                __builtin_prefetch(kbaseptr + (size_t)((kb + 1) * 32 + key) * HD + dseg, 0, 1);
            }
        }
        __syncthreads();

        if (kb * 32 <= qbase + 15) {     // wave-uniform causal skip
            // preload all K fragments from LDS, then run the WMMA chain
            v16bf kf[2][4];
            for (int nt = 0; nt < 2; ++nt)
                for (int kk = 0; kk < 4; ++kk)
                    kf[nt][kk] = load_frag(&kt[nt * 16][kk * 32], HD);
            v8f sc[2];
            for (int nt = 0; nt < 2; ++nt) {
                v8f s = {};
                for (int kk = 0; kk < 4; ++kk)
                    s = WMMA_BF16(qf[kk], kf[nt][kk], s);
                int col = kb * 32 + nt * 16 + nlo;
                for (int i = 0; i < 8; ++i) {
                    int row = qbase + half * 8 + i;
                    float v = s[i] * 0.08838834764831845f;   // 1/sqrt(128)
                    sc[nt][i] = (col > row) ? -1e30f : v;
                }
            }
            // online softmax: per-element index i <-> row (half*8+i)
            float rmax[8], rsum[8], alpha[8];
            for (int i = 0; i < 8; ++i) rmax[i] = fmaxf(sc[0][i], sc[1][i]);
            for (int m = 1; m <= 8; m <<= 1)
                for (int i = 0; i < 8; ++i)
                    rmax[i] = fmaxf(rmax[i], __shfl_xor(rmax[i], m, 32));
            for (int i = 0; i < 8; ++i) {
                float mn = fmaxf(mrow[i], rmax[i]);
                alpha[i] = __expf(mrow[i] - mn);
                mrow[i] = mn;
            }
            for (int nt = 0; nt < 2; ++nt)
                for (int i = 0; i < 8; ++i)
                    sc[nt][i] = __expf(sc[nt][i] - mrow[i]);
            for (int i = 0; i < 8; ++i) rsum[i] = sc[0][i] + sc[1][i];
            for (int m = 1; m <= 8; m <<= 1)
                for (int i = 0; i < 8; ++i)
                    rsum[i] += __shfl_xor(rsum[i], m, 32);
            for (int i = 0; i < 8; ++i) lrow[i] = lrow[i] * alpha[i] + rsum[i];
            for (int nt2 = 0; nt2 < 8; ++nt2)
                for (int i = 0; i < 8; ++i) o[nt2][i] *= alpha[i];

            // P (C-layout) -> LDS -> single 16x32 A fragment
            for (int nt = 0; nt < 2; ++nt)
                for (int i = 0; i < 8; ++i)
                    pls[wid][half * 8 + i][nt * 16 + nlo] = (__bf16)sc[nt][i];
            v16bf pf = load_frag(&pls[wid][0][0], 32);
            // preload all 8 V fragments, then run the PV WMMA chain
            v16bf vf[8];
            for (int nt2 = 0; nt2 < 8; ++nt2)
                vf[nt2] = load_frag(&vt[nt2 * 16][0], 32);
            for (int nt2 = 0; nt2 < 8; ++nt2)
                o[nt2] = WMMA_BF16(pf, vf[nt2], o[nt2]);
        }
    }

    float inv[8];
    for (int i = 0; i < 8; ++i) inv[i] = (lrow[i] > 0.0f) ? 1.0f / lrow[i] : 0.0f;
    __bf16* xb = X2 + ((size_t)b * SEQ + qbase + half * 8) * NQ + h * HD + nlo;
    for (int nt2 = 0; nt2 < 8; ++nt2)
        for (int i = 0; i < 8; ++i)
            xb[(size_t)i * NQ + nt2 * 16] = (__bf16)(o[nt2][i] * inv[i]);
}

// ---------------- host orchestration ----------------
extern "C" void kernel_launch(void* const* d_in, const int* in_sizes, int n_in,
                              void* d_out, int out_size, void* d_ws, size_t ws_size,
                              hipStream_t stream) {
    const float* hs  = (const float*)d_in[0];
    const float* Wq  = (const float*)d_in[1];
    const float* Wk  = (const float*)d_in[2];
    const float* Wv  = (const float*)d_in[3];
    const float* Wo  = (const float*)d_in[4];
    const int*   pid = (const int*)d_in[5];
    float* out = (float*)d_out;

    const size_t nX  = (size_t)MROWS * HIDD;   // 16.78M
    const size_t nWq = (size_t)NQ * HIDD;
    const size_t nWk = (size_t)NKVD * HIDD;
    const size_t nQ  = (size_t)MROWS * NQ;
    const size_t nK  = (size_t)MROWS * NKVD;

    unsigned char* w = (unsigned char*)d_ws;
    auto alloc = [&](size_t bytes) -> void* {
        void* p = (void*)w;
        w += (bytes + 255) & ~(size_t)255;
        return p;
    };
    __bf16* Xb   = (__bf16*)alloc(nX * 2);
    __bf16* Wqb  = (__bf16*)alloc(nWq * 2);
    __bf16* Wkb  = (__bf16*)alloc(nWk * 2);
    __bf16* Wvb  = (__bf16*)alloc(nWk * 2);
    __bf16* Wob  = (__bf16*)alloc(nWq * 2);
    float*  Qf   = (float*)alloc(nQ * 4);
    float*  Kf   = (float*)alloc(nK * 4);
    float*  Vf   = (float*)alloc(nK * 4);
    __bf16* Qb   = (__bf16*)alloc(nQ * 2);   // [B,NH,S,HD]
    __bf16* Kb   = (__bf16*)alloc(nK * 2);   // [B,NKV,S,HD]
    __bf16* Vb   = (__bf16*)alloc(nK * 2);   // [B,NKV,S,HD]
    __bf16* X2b  = (__bf16*)alloc(nQ * 2);   // [B*S, NH*HD]

    // 1) casts to bf16
    cast_f32_bf16<<<dim3((unsigned)((nX  + 255) / 256)), 256, 0, stream>>>(hs, Xb,  nX);
    cast_f32_bf16<<<dim3((unsigned)((nWq + 255) / 256)), 256, 0, stream>>>(Wq, Wqb, nWq);
    cast_f32_bf16<<<dim3((unsigned)((nWk + 255) / 256)), 256, 0, stream>>>(Wk, Wkb, nWk);
    cast_f32_bf16<<<dim3((unsigned)((nWk + 255) / 256)), 256, 0, stream>>>(Wv, Wvb, nWk);
    cast_f32_bf16<<<dim3((unsigned)((nWq + 255) / 256)), 256, 0, stream>>>(Wo, Wob, nWq);

    // 2) projections
    gemm_bf16nt<NQ,   HIDD><<<dim3(NQ / BNT,   MROWS / BMT), 256, 0, stream>>>(Xb, Wqb, Qf);
    gemm_bf16nt<NKVD, HIDD><<<dim3(NKVD / BNT, MROWS / BMT), 256, 0, stream>>>(Xb, Wkb, Kf);
    gemm_bf16nt<NKVD, HIDD><<<dim3(NKVD / BNT, MROWS / BMT), 256, 0, stream>>>(Xb, Wvb, Vf);

    // 3) RoPE + layout
    size_t ropeQ = (size_t)BATCH * SEQ * NH  * (HD / 2);
    size_t ropeK = (size_t)BATCH * SEQ * NKV * (HD / 2);
    size_t vres  = (size_t)BATCH * SEQ * NKV * HD;
    rope_kernel<<<dim3((unsigned)((ropeQ + 255) / 256)), 256, 0, stream>>>(Qf, pid, Qb, NH);
    rope_kernel<<<dim3((unsigned)((ropeK + 255) / 256)), 256, 0, stream>>>(Kf, pid, Kb, NKV);
    v_reshape  <<<dim3((unsigned)((vres  + 255) / 256)), 256, 0, stream>>>(Vf, Vb);

    // 4) attention
    attn_kernel<<<dim3(SEQ / 64, NH, BATCH), 128, 0, stream>>>(Qb, Kb, Vb, X2b);

    // 5) output projection -> d_out (f32)
    gemm_bf16nt<HIDD, NQ><<<dim3(HIDD / BNT, MROWS / BMT), 256, 0, stream>>>(X2b, Wob, out);

    (void)in_sizes; (void)n_in; (void)out_size; (void)ws_size;
}
